// QuantizedLinear_28260884808075
// MI455X (gfx1250) — matrix operand order repaired and checked by reading, not compiled, not run on hardware
//
#include <hip/hip_runtime.h>
#include <hip/hip_bf16.h>

typedef _Float16 v16h __attribute__((ext_vector_type(16)));
typedef _Float16 v8h  __attribute__((ext_vector_type(8)));
typedef _Float16 h4   __attribute__((ext_vector_type(4)));
typedef float    v8f  __attribute__((ext_vector_type(8)));

#define K_DIM 4096
#define N_DIM 11008
#define BLK_M 128
#define BLK_N 128
#define BLK_K 32
#define LDS_STRIDE 40   // 32 + 8 halves pad -> 80B row stride (gcd(20,64)=4) => conflict-free 16B frag reads

__global__ __launch_bounds__(256) void qlinear_wmma_f16(
    const float* __restrict__ x,      // [M, K] fp32
    const int*   __restrict__ wq,     // [N, K] int32 (int8-valued)
    const float* __restrict__ wscale, // [N]
    float*       __restrict__ out)    // [M, N]
{
    __shared__ _Float16 As[2][BLK_M][LDS_STRIDE];
    __shared__ _Float16 Bs[2][BLK_N][LDS_STRIDE];

    const int tid   = threadIdx.x;
    const int lane  = tid & 31;
    const int wave  = tid >> 5;
    const int waveM = wave & 3;    // 4 waves along M, 32 rows each
    const int waveN = wave >> 2;   // 2 waves along N, 64 cols each
    const int l16   = lane & 15;
    const int lhalf = lane >> 4;   // 0: lanes 0-15, 1: lanes 16-31

    const int blockN = blockIdx.x * BLK_N;
    const int blockM = blockIdx.y * BLK_M;

    // 256 threads: each loads 4x float4 (x) + 4x int4 (W) per K-step
    const int ldRow  = tid >> 3;   // 0..31 (row within 32-row pass)
    const int ldQuad = tid & 7;    // 0..7  (16-byte chunk within 32-wide K slab)

    const v8f vzero = {0.f, 0.f, 0.f, 0.f, 0.f, 0.f, 0.f, 0.f};
    v8f acc[2][4];
    #pragma unroll
    for (int mt = 0; mt < 2; ++mt)
        #pragma unroll
        for (int nt = 0; nt < 4; ++nt)
            acc[mt][nt] = vzero;

    float4 aReg[4];
    int4   bReg[4];

    // ---- initial fill of LDS buffer 0 (kt = 0) ----
    #pragma unroll
    for (int p = 0; p < 4; ++p) {
        const int r = ldRow + 32 * p;
        aReg[p] = *reinterpret_cast<const float4*>(x  + (size_t)(blockM + r) * K_DIM + ldQuad * 4);
        bReg[p] = *reinterpret_cast<const int4*>  (wq + (size_t)(blockN + r) * K_DIM + ldQuad * 4);
    }
    #pragma unroll
    for (int p = 0; p < 4; ++p) {
        const int r = ldRow + 32 * p;
        h4 ha = { (_Float16)aReg[p].x, (_Float16)aReg[p].y, (_Float16)aReg[p].z, (_Float16)aReg[p].w };
        h4 hb = { (_Float16)bReg[p].x, (_Float16)bReg[p].y, (_Float16)bReg[p].z, (_Float16)bReg[p].w };
        *reinterpret_cast<h4*>(&As[0][r][ldQuad * 4]) = ha;
        *reinterpret_cast<h4*>(&Bs[0][r][ldQuad * 4]) = hb;
    }
    __syncthreads();

    const int KT = K_DIM / BLK_K;  // 128 K-steps
    for (int kt = 0; kt < KT; ++kt) {
        const int cur = kt & 1;

        // global prefetch for next K-step (overlaps with WMMA compute below)
        if (kt + 1 < KT) {
            const int k0 = (kt + 1) * BLK_K;
            #pragma unroll
            for (int p = 0; p < 4; ++p) {
                const int r = ldRow + 32 * p;
                aReg[p] = *reinterpret_cast<const float4*>(x  + (size_t)(blockM + r) * K_DIM + k0 + ldQuad * 4);
                bReg[p] = *reinterpret_cast<const int4*>  (wq + (size_t)(blockN + r) * K_DIM + k0 + ldQuad * 4);
            }
        }
        // L2 prefetch two steps ahead
        if (kt + 2 < KT) {
            const int k2 = (kt + 2) * BLK_K;
            __builtin_prefetch(x  + (size_t)(blockM + ldRow) * K_DIM + k2, 0, 0);
            __builtin_prefetch(wq + (size_t)(blockN + ldRow) * K_DIM + k2, 0, 0);
        }

        // ---- gather A fragments (16x32 f16 layout, ISA 7.12.2) ----
        // lanes 0-15: row M=l16, K = {0..7} then {16..23}
        // lanes 16-31: row M=l16, K = {8..15} then {24..31}
        v16h afrag[2];
        #pragma unroll
        for (int mt = 0; mt < 2; ++mt) {
            const int row = waveM * 32 + mt * 16 + l16;
            const int ka  = lhalf ? 8 : 0;
            v8h c0 = *reinterpret_cast<const v8h*>(&As[cur][row][ka]);
            v8h c1 = *reinterpret_cast<const v8h*>(&As[cur][row][ka + 16]);
            #pragma unroll
            for (int i = 0; i < 8; ++i) { afrag[mt][i] = c0[i]; afrag[mt][8 + i] = c1[i]; }
        }
        // ---- gather B fragments (32x16 f16): col N=l16, lanes 0-15 K={0..7,16..23}, lanes 16-31 K={8..15,24..31} ----
        v16h bfrag[4];
        #pragma unroll
        for (int nt = 0; nt < 4; ++nt) {
            const int col = waveN * 64 + nt * 16 + l16;
            const int kb  = lhalf ? 8 : 0;
            v8h c0 = *reinterpret_cast<const v8h*>(&Bs[cur][col][kb]);
            v8h c1 = *reinterpret_cast<const v8h*>(&Bs[cur][col][kb + 16]);
            #pragma unroll
            for (int i = 0; i < 8; ++i) { bfrag[nt][i] = c0[i]; bfrag[nt][8 + i] = c1[i]; }
        }

        // ---- 8 WMMAs: 32x64 subtile per wave per K-step ----
        #pragma unroll
        for (int mt = 0; mt < 2; ++mt)
            #pragma unroll
            for (int nt = 0; nt < 4; ++nt)
                acc[mt][nt] = __builtin_amdgcn_wmma_f32_16x16x32_f16(
                    false, afrag[mt], false, bfrag[nt],
                    (short)0, acc[mt][nt], false, false);

        // ---- stage prefetched data into the other LDS buffer ----
        if (kt + 1 < KT) {
            __syncthreads();
            const int nxt = (kt + 1) & 1;
            #pragma unroll
            for (int p = 0; p < 4; ++p) {
                const int r = ldRow + 32 * p;
                h4 ha = { (_Float16)aReg[p].x, (_Float16)aReg[p].y, (_Float16)aReg[p].z, (_Float16)aReg[p].w };
                h4 hb = { (_Float16)bReg[p].x, (_Float16)bReg[p].y, (_Float16)bReg[p].z, (_Float16)bReg[p].w };
                *reinterpret_cast<h4*>(&As[nxt][r][ldQuad * 4]) = ha;
                *reinterpret_cast<h4*>(&Bs[nxt][r][ldQuad * 4]) = hb;
            }
            __syncthreads();
        }
    }

    // ---- epilogue: per-output-channel scale in f32, write D ----
    // C/D layout: VGPR r, lane l -> M = r + 8*(l>=16), N = l&15
    #pragma unroll
    for (int nt = 0; nt < 4; ++nt) {
        const int gCol = blockN + waveN * 64 + nt * 16 + l16;
        const float s  = wscale[gCol];
        #pragma unroll
        for (int mt = 0; mt < 2; ++mt) {
            const int rBase = blockM + waveM * 32 + mt * 16 + lhalf * 8;
            #pragma unroll
            for (int r = 0; r < 8; ++r) {
                out[(size_t)(rBase + r) * N_DIM + gCol] = acc[mt][nt][r] * s;
            }
        }
    }
}

extern "C" void kernel_launch(void* const* d_in, const int* in_sizes, int n_in,
                              void* d_out, int out_size, void* d_ws, size_t ws_size,
                              hipStream_t stream) {
    const float* x      = (const float*)d_in[0];
    const int*   wq     = (const int*)d_in[1];
    const float* wscale = (const float*)d_in[2];
    float*       out    = (float*)d_out;

    const int M = in_sizes[0] / K_DIM;  // 4*2048 = 8192
    dim3 grid(N_DIM / BLK_N, M / BLK_M);  // (86, 64)
    qlinear_wmma_f16<<<grid, 256, 0, stream>>>(x, wq, wscale, out);
}
